// LinearAttention_42992622633317
// MI455X (gfx1250) — matrix operand order repaired and checked by reading, not compiled, not run on hardware
//
#include <hip/hip_runtime.h>
#include <hip/hip_bf16.h>
#include <math.h>

// Shapes: x[4,4096,1024] f32, w_qkv[3072,1024] f32, out[4,16,4096,64] f32
// B=4 N=4096 C=1024 H=16 d=64, OUTC=3072, rows M = B*N = 16384.

typedef __attribute__((ext_vector_type(16))) __bf16       v16bf;
typedef __attribute__((ext_vector_type(4)))  __bf16       v4bf;
typedef __attribute__((ext_vector_type(8)))  float        v8f;
typedef __attribute__((ext_vector_type(4)))  unsigned int v4u;

union FragBF { v16bf v; v4u q[2]; };

#if defined(__AMDGCN__) && defined(__has_builtin)
#if __has_builtin(__builtin_amdgcn_global_load_async_to_lds_b128)
#define LA_USE_ASYNC 1
#endif
#endif
#ifndef LA_USE_ASYNC
#define LA_USE_ASYNC 0
#endif

#if LA_USE_ASYNC
// Builtin signature (from clang diagnostic): (v4i AS1*, v4i AS3*, imm int, imm int)
typedef int la_v4i __attribute__((vector_size(16)));
typedef __attribute__((address_space(1))) la_v4i as1_v4i;
typedef __attribute__((address_space(3))) la_v4i as3_v4i;
// Generic LDS pointer low 32 bits == LDS offset.
#define LA_TO_GLB(p) ((as1_v4i*)(unsigned long long)(p))
#define LA_TO_LDS(p) ((as3_v4i*)(unsigned int)(unsigned long long)(p))
#endif

// ---------------------------------------------------------------- f32 -> bf16
__global__ __launch_bounds__(256) void la_cvt_bf16(const float* __restrict__ in,
                                                   __bf16* __restrict__ out, int n4) {
  int i = blockIdx.x * 256 + threadIdx.x;
  if (i >= n4) return;
  float4 f = ((const float4*)in)[i];
  v4bf o;
  o.x = (__bf16)f.x; o.y = (__bf16)f.y; o.z = (__bf16)f.z; o.w = (__bf16)f.w;
  ((v4bf*)out)[i] = o;
}

// ---------------------------------------------------------------- zero scratch
__global__ __launch_bounds__(256) void la_zero_f32(float* p, int n) {
  int i = blockIdx.x * 256 + threadIdx.x;
  if (i < n) p[i] = 0.0f;
}

// ------------------------------------------------- QKV GEMM: [16384,3072,K=1024]
// qkv[m][o] = sum_k x[m][k] * w[o][k]   (both operands K-contiguous in memory,
// matching WMMA A-lane and B-lane packing read out of LDS below)
__global__ __launch_bounds__(256) void la_gemm_qkv(const __bf16* __restrict__ A,
                                                   const __bf16* __restrict__ Bw,
                                                   float* __restrict__ Cq) {
  constexpr int LDA = 40;                 // 32 K-halves padded to 40 (80B rows)
  const int tid = threadIdx.x;
  const int lane = tid & 31, wave = tid >> 5;
  const int waveM = wave & 3, waveN = wave >> 2;   // 4x2 wave grid
  const int m0 = blockIdx.y * 128, n0 = blockIdx.x * 128;
  const int l16 = lane & 15;
  const int hi  = (lane >= 16) ? 1 : 0;
  const int hi8 = hi ? 8 : 0;

  const v8f vz = {0.f,0.f,0.f,0.f,0.f,0.f,0.f,0.f};
  v8f acc[2][4];
  for (int mi = 0; mi < 2; ++mi)
    for (int ni = 0; ni < 4; ++ni) acc[mi][ni] = vz;

  // per-thread cooperative-copy chunk coordinates (2 x 16B per matrix tile)
  const int r0 = (tid + 0)   >> 2, s0 = (tid + 0)   & 3;
  const int r1 = (tid + 256) >> 2, s1 = (tid + 256) & 3;

#if LA_USE_ASYNC
  // ---------------- async double-buffered pipeline (ASYNCcnt-tracked copies)
  __shared__ __bf16 As[2][128 * LDA];
  __shared__ __bf16 Bs[2][128 * LDA];

  auto issue = [&](int buf, int k0) {
    __builtin_amdgcn_global_load_async_to_lds_b128(
        LA_TO_GLB(A + (long)(m0 + r0) * 1024 + k0 + s0 * 8),
        LA_TO_LDS(&As[buf][r0 * LDA + s0 * 8]), 0, 0);
    __builtin_amdgcn_global_load_async_to_lds_b128(
        LA_TO_GLB(Bw + (long)(n0 + r0) * 1024 + k0 + s0 * 8),
        LA_TO_LDS(&Bs[buf][r0 * LDA + s0 * 8]), 0, 0);
    __builtin_amdgcn_global_load_async_to_lds_b128(
        LA_TO_GLB(A + (long)(m0 + r1) * 1024 + k0 + s1 * 8),
        LA_TO_LDS(&As[buf][r1 * LDA + s1 * 8]), 0, 0);
    __builtin_amdgcn_global_load_async_to_lds_b128(
        LA_TO_GLB(Bw + (long)(n0 + r1) * 1024 + k0 + s1 * 8),
        LA_TO_LDS(&Bs[buf][r1 * LDA + s1 * 8]), 0, 0);
  };

  auto compute = [&](int buf) {
    const __bf16* Asb = &As[buf][0];
    const __bf16* Bsb = &Bs[buf][0];
    FragBF af[2], bfr[4];
    #pragma unroll
    for (int mi = 0; mi < 2; ++mi) {
      // A 16x32 bf16: lanes 0-15 hold K {0..7,16..23}; lanes 16-31 {8..15,24..31}
      int row = waveM * 32 + mi * 16 + l16;
      int ka = hi ? 8 : 0;
      af[mi].q[0] = *(const v4u*)(Asb + row * LDA + ka);
      af[mi].q[1] = *(const v4u*)(Asb + row * LDA + ka + 16);
    }
    #pragma unroll
    for (int ni = 0; ni < 4; ++ni) {
      // B 32x16 bf16: lanes 0-15 hold K {0..7,16..23}; lanes 16-31 {8..15,24..31}
      int col = waveN * 64 + ni * 16 + l16;
      int kb = hi ? 8 : 0;
      bfr[ni].q[0] = *(const v4u*)(Bsb + col * LDA + kb);
      bfr[ni].q[1] = *(const v4u*)(Bsb + col * LDA + kb + 16);
    }
    #pragma unroll
    for (int mi = 0; mi < 2; ++mi)
      #pragma unroll
      for (int ni = 0; ni < 4; ++ni)
        acc[mi][ni] = __builtin_amdgcn_wmma_f32_16x16x32_bf16(
            false, af[mi].v, false, bfr[ni].v, (short)0, acc[mi][ni], false, false);
  };

  issue(0, 0);
  // unroll 1: a single loop body keeps WMMA accumulators tied in place (no
  // cross-iteration register rotation -> no v_mov/v_nop hazard filler)
  #pragma unroll 1
  for (int kt = 0; kt < 31; ++kt) {
    issue((kt + 1) & 1, (kt + 1) * 32);
    // 4 outstanding ops belong to tile kt+1 => tile kt's copies are complete
    asm volatile("s_wait_asynccnt 0x4" ::: "memory");
    __syncthreads();
    compute(kt & 1);
    __syncthreads();  // all frag reads done before buffer (kt&1) is re-issued
  }
  asm volatile("s_wait_asynccnt 0x0" ::: "memory");
  __syncthreads();
  compute(1);

#else
  // ---------------- fallback: synchronous VGPR-staged copies
  __shared__ __bf16 As[128 * LDA];
  __shared__ __bf16 Bs[128 * LDA];

  for (int k0 = 0; k0 < 1024; k0 += 32) {
    *(v4u*)(As + r0 * LDA + s0 * 8) =
        *(const v4u*)(A + (long)(m0 + r0) * 1024 + k0 + s0 * 8);
    *(v4u*)(Bs + r0 * LDA + s0 * 8) =
        *(const v4u*)(Bw + (long)(n0 + r0) * 1024 + k0 + s0 * 8);
    *(v4u*)(As + r1 * LDA + s1 * 8) =
        *(const v4u*)(A + (long)(m0 + r1) * 1024 + k0 + s1 * 8);
    *(v4u*)(Bs + r1 * LDA + s1 * 8) =
        *(const v4u*)(Bw + (long)(n0 + r1) * 1024 + k0 + s1 * 8);
    __syncthreads();

    FragBF af[2], bfr[4];
    #pragma unroll
    for (int mi = 0; mi < 2; ++mi) {
      int row = waveM * 32 + mi * 16 + l16;
      int ka = hi ? 8 : 0;
      af[mi].q[0] = *(const v4u*)(As + row * LDA + ka);
      af[mi].q[1] = *(const v4u*)(As + row * LDA + ka + 16);
    }
    #pragma unroll
    for (int ni = 0; ni < 4; ++ni) {
      int col = waveN * 64 + ni * 16 + l16;
      int kb = hi ? 8 : 0;
      bfr[ni].q[0] = *(const v4u*)(Bs + col * LDA + kb);
      bfr[ni].q[1] = *(const v4u*)(Bs + col * LDA + kb + 16);
    }
    #pragma unroll
    for (int mi = 0; mi < 2; ++mi)
      #pragma unroll
      for (int ni = 0; ni < 4; ++ni)
        acc[mi][ni] = __builtin_amdgcn_wmma_f32_16x16x32_bf16(
            false, af[mi].v, false, bfr[ni].v, (short)0, acc[mi][ni], false, false);
    __syncthreads();
  }
#endif

  // C 16x16 f32: VGPR r holds M=r (lanes 0-15) / M=r+8 (lanes 16-31), N = lane%16
  #pragma unroll
  for (int mi = 0; mi < 2; ++mi)
    #pragma unroll
    for (int ni = 0; ni < 4; ++ni)
      #pragma unroll
      for (int r = 0; r < 8; ++r) {
        int row = m0 + waveM * 32 + mi * 16 + r + hi8;
        int col = n0 + waveN * 64 + ni * 16 + l16;
        Cq[(long)row * 3072 + col] = acc[mi][ni][r];
      }
}

// ------------------------------------- q softmax over d=64, *0.125, -> bf16 [bh][n][d]
__global__ __launch_bounds__(256) void la_q_softmax(const float* __restrict__ qkv,
                                                    __bf16* __restrict__ qb) {
  int wave = threadIdx.x >> 5, lane = threadIdx.x & 31;
  long row = (long)blockIdx.x * 8 + wave;      // (b*4096+n)*16 + h
  int h = (int)(row & 15);
  long bn = row >> 4;                          // b*4096+n
  const float* src = qkv + bn * 3072 + (long)h * 64;
  float x0 = src[lane], x1 = src[lane + 32];
  float m = fmaxf(x0, x1);
  for (int o = 16; o; o >>= 1) m = fmaxf(m, __shfl_xor(m, o, 32));
  float e0 = __expf(x0 - m), e1 = __expf(x1 - m);
  float s = e0 + e1;
  for (int o = 16; o; o >>= 1) s += __shfl_xor(s, o, 32);
  float inv = 0.125f / s;                      // d^-0.5 = 1/8
  int b = (int)(bn >> 12), n = (int)(bn & 4095);
  long obase = (((long)(b * 16 + h)) * 4096 + n) * 64;
  qb[obase + lane]      = (__bf16)(e0 * inv);
  qb[obase + lane + 32] = (__bf16)(e1 * inv);
}

// --------------------------------- k softmax over sequence dim (in place, fp32)
__global__ __launch_bounds__(256) void la_k_softmax(float* qkv) {
  int bh = blockIdx.x;                 // b*16+h
  int b = bh >> 4, h = bh & 15;
  int dd = threadIdx.x & 63, g = threadIdx.x >> 6;
  float* base = qkv + (long)b * 4096 * 3072 + 1024 + (long)h * 64 + dd;
  __shared__ float red[4][64];
  float m = -3.0e38f;
  for (int i = g; i < 4096; i += 4) m = fmaxf(m, base[(long)i * 3072]);
  red[g][dd] = m;
  __syncthreads();
  m = fmaxf(fmaxf(red[0][dd], red[1][dd]), fmaxf(red[2][dd], red[3][dd]));
  __syncthreads();
  float s = 0.0f;
  for (int i = g; i < 4096; i += 4) s += __expf(base[(long)i * 3072] - m);
  red[g][dd] = s;
  __syncthreads();
  s = red[0][dd] + red[1][dd] + red[2][dd] + red[3][dd];
  float inv = 1.0f / s;
  for (int i = g; i < 4096; i += 4)
    base[(long)i * 3072] = __expf(base[(long)i * 3072] - m) * inv;
}

// --------------------- context^T[bh][e][dd] = sum_n v[n][e]*k[n][dd], split-K atomics
__global__ __launch_bounds__(256) void la_context(const float* __restrict__ qkv,
                                                  float* __restrict__ ctxT) {
  int z = blockIdx.x;                  // 64 bh * 8 K-chunks
  int bh = z >> 3, chunk = z & 7;
  int b = bh >> 4, h = bh & 15;
  __shared__ float kS[32][64];
  __shared__ float vS[32][64];
  int tid = threadIdx.x;
  int dd = tid & 63, eg = tid >> 6;
  float acc[16];
  #pragma unroll
  for (int i = 0; i < 16; ++i) acc[i] = 0.0f;
  const float* kb = qkv + (long)b * 4096 * 3072 + 1024 + (long)h * 64;
  const float* vb = kb + 1024;
  for (int s = 0; s < 16; ++s) {
    int n0 = chunk * 512 + s * 32;
    #pragma unroll
    for (int j = 0; j < 8; ++j) {
      int f = tid + 256 * j;
      int r = f >> 6, c = f & 63;
      kS[r][c] = kb[(long)(n0 + r) * 3072 + c];
      vS[r][c] = vb[(long)(n0 + r) * 3072 + c];
    }
    __syncthreads();
    for (int nn = 0; nn < 32; ++nn) {
      float kv = kS[nn][dd];
      #pragma unroll
      for (int i = 0; i < 16; ++i) acc[i] += kv * vS[nn][eg * 16 + i];
    }
    __syncthreads();
  }
  #pragma unroll
  for (int i = 0; i < 16; ++i) {
    int e = eg * 16 + i;
    atomicAdd(&ctxT[(long)bh * 4096 + e * 64 + dd], acc[i]);
  }
}

// ----------------------- out[bh][n][e] = sum_dd q[bh][n][dd]*ctxT[bh][e][dd] (WMMA)
__global__ __launch_bounds__(256) void la_out_gemm(const __bf16* __restrict__ qb,
                                                   const __bf16* __restrict__ ctxb,
                                                   float* __restrict__ out) {
  constexpr int LDC = 72;              // 64 halves padded to 72 (144B rows)
  __shared__ __bf16 Cs[64 * LDC];
  int bh = blockIdx.x;
  int m0 = blockIdx.y * 128;
  int tid = threadIdx.x, lane = tid & 31, wave = tid >> 5;
  int l16 = lane & 15;
  int hi = (lane >= 16) ? 1 : 0;
  int hi8 = hi ? 8 : 0;

  const __bf16* csrc = ctxb + (long)bh * 4096;     // 64x64 bf16 = 512 x 16B chunks
  #pragma unroll
  for (int j = 0; j < 2; ++j) {
    int c = tid + 256 * j;
    int r = c >> 3, s = c & 7;
    *(v4u*)(Cs + r * LDC + s * 8) = *(const v4u*)(csrc + r * 64 + s * 8);
  }
  __syncthreads();

  const v8f vz = {0.f,0.f,0.f,0.f,0.f,0.f,0.f,0.f};
  v8f acc[4] = {vz, vz, vz, vz};
  const __bf16* qbase = qb + ((long)bh * 4096 + m0 + wave * 16) * 64;
  #pragma unroll
  for (int ks = 0; ks < 2; ++ks) {
    FragBF af;
    int ka = ks * 32 + (hi ? 8 : 0);
    af.q[0] = *(const v4u*)(qbase + (long)l16 * 64 + ka);
    af.q[1] = *(const v4u*)(qbase + (long)l16 * 64 + ka + 16);
    #pragma unroll
    for (int ni = 0; ni < 4; ++ni) {
      FragBF bfr;
      int e = ni * 16 + l16;
      int kbo = ks * 32 + (hi ? 8 : 0);
      bfr.q[0] = *(const v4u*)(Cs + e * LDC + kbo);
      bfr.q[1] = *(const v4u*)(Cs + e * LDC + kbo + 16);
      acc[ni] = __builtin_amdgcn_wmma_f32_16x16x32_bf16(
          false, af.v, false, bfr.v, (short)0, acc[ni], false, false);
    }
  }
  #pragma unroll
  for (int ni = 0; ni < 4; ++ni)
    #pragma unroll
    for (int r = 0; r < 8; ++r) {
      int row = m0 + wave * 16 + r + hi8;
      int col = ni * 16 + l16;
      out[((long)bh * 4096 + row) * 64 + col] = acc[ni][r];
    }
}

// ---------------------------------------------------------------- launcher
extern "C" void kernel_launch(void* const* d_in, const int* in_sizes, int n_in,
                              void* d_out, int out_size, void* d_ws, size_t ws_size,
                              hipStream_t stream) {
  const float* x = (const float*)d_in[0];   // [4,4096,1024]
  const float* w = (const float*)d_in[1];   // [3072,1024]
  float* out = (float*)d_out;               // [4,16,4096,64]
  char* ws = (char*)d_ws;

  // workspace layout (~276 MB total)
  __bf16* xb   = (__bf16*)(ws);                       // 33,554,432 B
  __bf16* wb   = (__bf16*)(ws + 33554432u);           //  6,291,456 B
  float*  qkv  = (float*) (ws + 39845888u);           // 201,326,592 B
  __bf16* qb   = (__bf16*)(ws + 241172480u);          // 33,554,432 B
  float*  ctxf = (float*) (ws + 274726912u);          //  1,048,576 B
  __bf16* ctxb = (__bf16*)(ws + 275775488u);          //    524,288 B

  la_cvt_bf16<<<16384, 256, 0, stream>>>(x, xb, 16384 * 1024 / 4);
  la_cvt_bf16<<<3072, 256, 0, stream>>>(w, wb, 3072 * 1024 / 4);
  la_gemm_qkv<<<dim3(24, 128), 256, 0, stream>>>(xb, wb, qkv);
  la_q_softmax<<<32768, 256, 0, stream>>>(qkv, qb);
  la_zero_f32<<<1024, 256, 0, stream>>>(ctxf, 262144);
  la_k_softmax<<<64, 256, 0, stream>>>(qkv);
  la_context<<<512, 256, 0, stream>>>(qkv, ctxf);
  la_cvt_bf16<<<256, 256, 0, stream>>>(ctxf, ctxb, 262144 / 4);
  la_out_gemm<<<dim3(64, 32), 256, 0, stream>>>(qb, ctxb, out);
}
